// LineLayer_3917010174431
// MI455X (gfx1250) — matrix operand order repaired and checked by reading, not compiled, not run on hardware
//
#include <hip/hip_runtime.h>
#include <hip/hip_bf16.h>

typedef __attribute__((ext_vector_type(16))) __bf16 v16bf;
typedef __attribute__((ext_vector_type(8)))  __bf16 v8bf;
typedef __attribute__((ext_vector_type(8)))  float  v8f;

#define B_  8
#define D_  256
#define N_  8192
#define M_  8192
#define K1_ 768
#define O1_ 512
#define K2_ 512
#define O2_ 256

static __device__ __forceinline__ unsigned short f2bf(float f) {
  return __builtin_bit_cast(unsigned short, (__bf16)f);
}
static __device__ __forceinline__ float bf2f(unsigned short u) {
  return (float)__builtin_bit_cast(__bf16, u);
}

// ---------------------------------------------------------------- zero
__global__ void zero_kernel(float* p, long n) {
  long i = blockIdx.x * (long)blockDim.x + threadIdx.x;
  long stride = (long)gridDim.x * blockDim.x;
  for (; i < n; i += stride) p[i] = 0.0f;
}

// ---------------------------------------------------------------- weights -> bf16
__global__ void cvt_w_kernel(const float* __restrict__ W1, const float* __restrict__ W2,
                             unsigned short* __restrict__ w1b, unsigned short* __restrict__ w2b) {
  int i = blockIdx.x * blockDim.x + threadIdx.x;
  const int n1 = O1_ * K1_;
  const int total = n1 + O2_ * K2_;
  if (i < n1) w1b[i] = f2bf(W1[i]);
  else if (i < total) w2b[i - n1] = f2bf(W2[i - n1]);
}

// ---------------------------------------------------------------- junction hit counts
__global__ void count_kernel(const int* __restrict__ idx0, const int* __restrict__ idx1,
                             float* __restrict__ cnt) {
  int t = blockIdx.x * blockDim.x + threadIdx.x;    // 2*B*M threads
  int img = t >> 16;
  int rem = t & 65535;                               // b*M + m
  const int* idx = img ? idx1 : idx0;
  int n = idx[rem];
  atomicAdd(&cnt[(img << 16) + ((rem >> 13) << 13) + n], 1.0f);
}

// ---------------------------------------------------------------- GEMM1: h = W1 @ msg + b1 (pre-BN), bf16 out + stats
// Block tile: 64 m x 128 o. 8 waves: wave_m in 0..3 (16 m each), wave_o in 0..1 (64 o each, 4 WMMA tiles).
// Double-buffered LDS A-tile; region of msg (gather / neighbor / line_enc) is uniform per K-step (kk>>8).
__global__ __launch_bounds__(256) void gemm1_kernel(
    const float* __restrict__ ldesc0, const float* __restrict__ ldesc1,
    const float* __restrict__ le0, const float* __restrict__ le1,
    const int* __restrict__ idx0, const int* __restrict__ idx1,
    const unsigned short* __restrict__ w1b, const float* __restrict__ b1,
    unsigned short* __restrict__ hbuf, float* __restrict__ gsum, float* __restrict__ gsumsq) {
  __shared__ __bf16 lds_a[2][64 * 40];   // 2 x (64 m rows, 32 k + 8 pad)

  const int tid = threadIdx.x;
  const int lane = tid & 31, wave = tid >> 5;
  const int wave_m = wave & 3, wave_o = wave >> 2;
  const int bz = blockIdx.z, img = bz >> 3, b = bz & 7;

  const float* ldb = (img ? ldesc1 : ldesc0) + (size_t)b * D_ * N_;
  const float* leb = (img ? le1 : le0) + (size_t)b * D_ * M_;
  const int* jidx = (img ? idx1 : idx0) + b * M_;

  const int m_block = blockIdx.x * 64;
  const int o_block = blockIdx.y * 128;

  // A-tile staging mapping: each thread fills msg^T[m_local][sk..sk+7]
  const int sm = tid & 63;
  const int sk = (tid >> 6) << 3;
  const int m_g = m_block + sm;
  const int j0 = jidx[m_g];
  const int j1 = jidx[m_g ^ 1];   // paired endpoint (reshape(..,2)[..,::-1])

  auto stage = [&](int kk, __bf16* dst) {
    const int r = kk >> 8;              // UNIFORM across block (kk mult of 32, sk<32)
    const int kl = (kk & 255) + sk;     // k within region
    const float* src;
    if (r == 0)      src = ldb + (size_t)kl * 8192 + j0;   // gathered endpoint
    else if (r == 1) src = ldb + (size_t)kl * 8192 + j1;   // neighbor endpoint
    else             src = leb + (size_t)kl * 8192 + m_g;  // line encoding
    alignas(16) unsigned short tmp[8];
#pragma unroll
    for (int j = 0; j < 8; ++j) tmp[j] = f2bf(src[(size_t)j * 8192]);
    *reinterpret_cast<v8bf*>(dst + sm * 40 + sk) = *reinterpret_cast<const v8bf*>(tmp);
  };

  const int row = lane & 15, hi = lane >> 4;
  const int k0 = hi * 16;
  const int arow_off = (wave_m * 16 + row) * 40;
  const int o_base = o_block + wave_o * 64;
  const unsigned short* bp[4];
#pragma unroll
  for (int t = 0; t < 4; ++t)
    bp[t] = w1b + (size_t)(o_base + t * 16 + row) * K1_ + hi * 16;

  v8f acc[4];
#pragma unroll
  for (int t = 0; t < 4; ++t) acc[t] = (v8f){};

  stage(0, lds_a[0]);
  int buf = 0;
  for (int kk = 0; kk < K1_; kk += 32, buf ^= 1) {
    __syncthreads();
    if (kk + 32 < K1_) stage(kk + 32, lds_a[buf ^ 1]);   // overlap next gather with WMMA

    __builtin_prefetch(bp[0] + kk + 32, 0, 0);
    const __bf16* arow = lds_a[buf] + arow_off;
    union { v16bf v; v8bf h[2]; } a;
    a.h[0] = *reinterpret_cast<const v8bf*>(arow + k0);
    a.h[1] = *reinterpret_cast<const v8bf*>(arow + k0 + 8);
#pragma unroll
    for (int t = 0; t < 4; ++t) {
      v16bf bf = *reinterpret_cast<const v16bf*>(bp[t] + kk);
      acc[t] = __builtin_amdgcn_wmma_f32_16x16x32_bf16(false, a.v, false, bf, (short)0, acc[t],
                                                       false, false);
    }
  }

  // Epilogue: +b1, per-channel stats, bf16 store of h (lane holds fixed o, 8 consecutive m)
  const int m_base = m_block + wave_m * 16 + hi * 8;
  const size_t hb_base = (size_t)bz * O1_ * M_;
#pragma unroll
  for (int t = 0; t < 4; ++t) {
    const int o_g = o_base + t * 16 + row;
    const float bias = b1[o_g];
    float s = 0.0f, s2 = 0.0f;
    alignas(16) unsigned short pack[8];
#pragma unroll
    for (int r = 0; r < 8; ++r) {
      float v = acc[t][r] + bias;
      s += v; s2 += v * v;
      pack[r] = f2bf(v);
    }
    s  += __shfl_xor(s, 16, 32);
    s2 += __shfl_xor(s2, 16, 32);
    if (lane < 16) {
      atomicAdd(&gsum[img * O1_ + o_g], s);
      atomicAdd(&gsumsq[img * O1_ + o_g], s2);
    }
    *reinterpret_cast<v8bf*>(hbuf + hb_base + (size_t)o_g * M_ + m_base) =
        *reinterpret_cast<const v8bf*>(pack);
  }
}

// ---------------------------------------------------------------- fold BN stats into scale/shift
__global__ void stats_kernel(const float* __restrict__ gsum, const float* __restrict__ gsumsq,
                             const float* __restrict__ g1, const float* __restrict__ bt1,
                             float* __restrict__ scale, float* __restrict__ shift) {
  int t = blockIdx.x * blockDim.x + threadIdx.x;
  if (t >= 2 * O1_) return;
  int ch = t & (O1_ - 1);
  const float inv = 1.0f / (float)(B_ * M_);
  float mean = gsum[t] * inv;
  float var = gsumsq[t] * inv - mean * mean;
  var = var < 0.0f ? 0.0f : var;
  float sc = g1[ch] * rsqrtf(var + 1e-5f);
  scale[t] = sc;
  shift[t] = bt1[ch] - mean * sc;
}

// ---------------------------------------------------------------- GEMM2: up = W2 @ relu(bn(h)) + b2, scatter-add
// Block tile: 64 m x 128 d. 8 waves: wave_m in 0..3, wave_o in 0..1 (4 WMMA tiles each). Double-buffered LDS.
__global__ __launch_bounds__(256) void gemm2_kernel(
    const unsigned short* __restrict__ hbuf, const unsigned short* __restrict__ w2b,
    const float* __restrict__ scale, const float* __restrict__ shift,
    const float* __restrict__ b2,
    const int* __restrict__ idx0, const int* __restrict__ idx1,
    float* __restrict__ ssum) {
  __shared__ __bf16 lds_a[2][64 * 40];

  const int tid = threadIdx.x;
  const int lane = tid & 31, wave = tid >> 5;
  const int wave_m = wave & 3, wave_o = wave >> 2;
  const int bz = blockIdx.z, img = bz >> 3, b = bz & 7;

  const int* jidx = (img ? idx1 : idx0) + b * M_;
  const int m_block = blockIdx.x * 64;
  const int d_block = blockIdx.y * 128;
  const unsigned short* hb = hbuf + (size_t)bz * K2_ * M_;

  // staging: thread handles k_local = tid&31, 8 consecutive m (coalesced load, transposed LDS store)
  const int sk = tid & 31;
  const int smc = (tid >> 5) << 3;

  auto stage = [&](int kk, __bf16* dst) {
    const int kg = kk + sk;
    const float sc = scale[img * O1_ + kg];
    const float sh = shift[img * O1_ + kg];
    alignas(16) unsigned short raw[8];
    *reinterpret_cast<v8bf*>(raw) =
        *reinterpret_cast<const v8bf*>(hb + (size_t)kg * M_ + m_block + smc);
#pragma unroll
    for (int j = 0; j < 8; ++j) {
      float v = bf2f(raw[j]) * sc + sh;        // BatchNorm affine
      v = v > 0.0f ? v : 0.0f;                 // ReLU
      dst[(smc + j) * 40 + sk] = (__bf16)v;    // transpose into [m][k]
    }
  };

  const int row = lane & 15, hi = lane >> 4;
  const int k0 = hi * 16;
  const int arow_off = (wave_m * 16 + row) * 40;
  const int d_base = d_block + wave_o * 64;
  const unsigned short* bp[4];
#pragma unroll
  for (int t = 0; t < 4; ++t)
    bp[t] = w2b + (size_t)(d_base + t * 16 + row) * K2_ + hi * 16;

  v8f acc[4];
#pragma unroll
  for (int t = 0; t < 4; ++t) acc[t] = (v8f){};

  stage(0, lds_a[0]);
  int buf = 0;
  for (int kk = 0; kk < K2_; kk += 32, buf ^= 1) {
    __syncthreads();
    if (kk + 32 < K2_) stage(kk + 32, lds_a[buf ^ 1]);

    __builtin_prefetch(bp[0] + kk + 32, 0, 0);
    const __bf16* arow = lds_a[buf] + arow_off;
    union { v16bf v; v8bf h[2]; } a;
    a.h[0] = *reinterpret_cast<const v8bf*>(arow + k0);
    a.h[1] = *reinterpret_cast<const v8bf*>(arow + k0 + 8);
#pragma unroll
    for (int t = 0; t < 4; ++t) {
      v16bf bf = *reinterpret_cast<const v16bf*>(bp[t] + kk);
      acc[t] = __builtin_amdgcn_wmma_f32_16x16x32_bf16(false, a.v, false, bf, (short)0, acc[t],
                                                       false, false);
    }
  }

  const int m_base = m_block + wave_m * 16 + hi * 8;
  float* sb = ssum + (size_t)bz * O2_ * N_;
#pragma unroll
  for (int t = 0; t < 4; ++t) {
    const int d_g = d_base + t * 16 + row;
    const float bias = b2[d_g];
#pragma unroll
    for (int r = 0; r < 8; ++r) {
      const int m = m_base + r;
      const int n = jidx[m];
      atomicAdd(&sb[(size_t)d_g * N_ + n], acc[t][r] + bias);
    }
  }
}

// ---------------------------------------------------------------- out = ldesc + ssum / max(cnt,1)
__global__ void final_kernel(const float* __restrict__ ldesc0, const float* __restrict__ ldesc1,
                             const float* __restrict__ ssum, const float* __restrict__ cnt,
                             float* __restrict__ out) {
  long t = blockIdx.x * (long)blockDim.x + threadIdx.x;
  const long total = 2L * B_ * D_ * N_;
  const long stride = (long)gridDim.x * blockDim.x;
  for (; t < total; t += stride) {
    long n = t & (N_ - 1);
    long rest = t >> 13;
    long d = rest & (D_ - 1);
    long rest2 = rest >> 8;
    long b = rest2 & 7;
    long img = rest2 >> 3;
    const float* ld = img ? ldesc1 : ldesc0;
    float c = cnt[((img * 8 + b) << 13) + n];
    c = c > 1.0f ? c : 1.0f;
    out[t] = ld[((b << 8) + d) * N_ + n] + ssum[t] / c;
  }
}

// ---------------------------------------------------------------- launch
extern "C" void kernel_launch(void* const* d_in, const int* in_sizes, int n_in,
                              void* d_out, int out_size, void* d_ws, size_t ws_size,
                              hipStream_t stream) {
  const float* ldesc0 = (const float*)d_in[0];
  const float* ldesc1 = (const float*)d_in[1];
  const float* le0    = (const float*)d_in[2];
  const float* le1    = (const float*)d_in[3];
  const int*   idx0   = (const int*)d_in[4];
  const int*   idx1   = (const int*)d_in[5];
  const float* W1     = (const float*)d_in[6];
  const float* b1     = (const float*)d_in[7];
  const float* g1     = (const float*)d_in[8];
  const float* bt1    = (const float*)d_in[9];
  const float* W2     = (const float*)d_in[10];
  const float* b2     = (const float*)d_in[11];
  float* out = (float*)d_out;

  char* ws = (char*)d_ws;
  float* gsum   = (float*)(ws + 0);            // 2*512 f32
  float* gsumsq = (float*)(ws + 4096);         // 2*512 f32
  float* scale  = (float*)(ws + 8192);         // 2*512 f32
  float* shift  = (float*)(ws + 12288);        // 2*512 f32
  float* cnt    = (float*)(ws + 16384);        // 2*8*8192 f32
  float* ssum   = (float*)(ws + 540672);       // 2*8*256*8192 f32  (128 MB)
  unsigned short* hbuf = (unsigned short*)(ws + 134758400); // 2*8*512*8192 bf16 (128 MB)
  unsigned short* w1b  = (unsigned short*)(ws + 268976128); // 512*768 bf16
  unsigned short* w2b  = (unsigned short*)(ws + 269762560); // 256*512 bf16

  const long nz = (540672 + 134217728) / 4;    // stats + cnt + ssum
  zero_kernel<<<4096, 256, 0, stream>>>(gsum, nz);
  cvt_w_kernel<<<(O1_ * K1_ + O2_ * K2_ + 255) / 256, 256, 0, stream>>>(W1, W2, w1b, w2b);
  count_kernel<<<(2 * B_ * M_) / 256, 256, 0, stream>>>(idx0, idx1, cnt);
  gemm1_kernel<<<dim3(M_ / 64, O1_ / 128, 2 * B_), 256, 0, stream>>>(
      ldesc0, ldesc1, le0, le1, idx0, idx1, w1b, b1, hbuf, gsum, gsumsq);
  stats_kernel<<<4, 256, 0, stream>>>(gsum, gsumsq, g1, bt1, scale, shift);
  gemm2_kernel<<<dim3(M_ / 64, O2_ / 128, 2 * B_), 256, 0, stream>>>(
      hbuf, w2b, scale, shift, b2, idx0, idx1, ssum);
  final_kernel<<<4096, 256, 0, stream>>>(ldesc0, ldesc1, ssum, cnt, out);
}
